// DGCNNLwf3Dv2_2_46703474377491
// MI455X (gfx1250) — matrix operand order repaired and checked by reading, not compiled, not run on hardware
//
#include <hip/hip_runtime.h>
#include <hip/hip_bf16.h>

// ---------------------------------------------------------------------------
// DGCNN-LwF forward for MI455X (gfx1250, wave32, WMMA bf16 16x16x32)
//  - weights pre-swizzled into WMMA A-fragment lane layout (zero-padded K)
//  - edge features staged per column-tile into LDS in B-fragment lane layout
//  - guard-free 32B vector loads feeding v_wmma_f32_16x16x32_bf16
// ---------------------------------------------------------------------------

#define Bc 8
#define Nc 2048
#define Kc 20

typedef __attribute__((ext_vector_type(16))) __bf16 v16bf;
typedef __attribute__((ext_vector_type(8)))  float  v8f;

__device__ __forceinline__ unsigned short f2bfu(float f) {
    unsigned u = __float_as_uint(f);
    u += 0x7fffu + ((u >> 16) & 1u);          // round-to-nearest-even
    return (unsigned short)(u >> 16);
}
// order-preserving float<->uint encoding (atomic max on floats incl. negatives)
__device__ __forceinline__ unsigned f2ord(float f) {
    unsigned u = __float_as_uint(f);
    return (u & 0x80000000u) ? ~u : (u | 0x80000000u);
}
__device__ __forceinline__ float ord2f(unsigned u) {
    return __uint_as_float((u & 0x80000000u) ? (u & 0x7fffffffu) : ~u);
}

// ---------------------------------------------------------------------------
// prep kernels
// ---------------------------------------------------------------------------
__global__ void k_transpose_x(const float* __restrict__ x, float* __restrict__ xt) {
    int i = blockIdx.x * 256 + threadIdx.x;       // over B*N
    if (i >= Bc * Nc) return;
    int b = i / Nc, n = i % Nc;
    for (int c = 0; c < 3; ++c)
        xt[((size_t)b * Nc + n) * 3 + c] = x[((size_t)b * 3 + c) * Nc + n];
}

// Swizzle W[O][C2] (f32) into WMMA A-fragment order, K padded to NKC*32:
// dst[((ot*NKC+kc)*32 + lane)*16 + i] = W[ot*16 + (lane&15)][kc*32 + kla(i,half)]
__global__ void k_wlayout(const float* __restrict__ w, unsigned short* __restrict__ d,
                          int O, int C2, int nkc) {
    int e = blockIdx.x * 256 + threadIdx.x;
    int total = (O >> 4) * nkc * 512;
    if (e >= total) return;
    int i    = e & 15;
    int lane = (e >> 4) & 31;
    int kc   = (e >> 9) % nkc;
    int ot   = e / (nkc << 9);
    int half = lane >> 4, ln = lane & 15;
    int kla  = half * 16 + i;
    int K    = kc * 32 + kla;
    d[e] = (K < C2) ? f2bfu(w[(size_t)(ot * 16 + ln) * C2 + K]) : (unsigned short)0;
}

__global__ void k_bn_affine(const float* __restrict__ p, float2* __restrict__ aff, int C) {
    int i = blockIdx.x * 256 + threadIdx.x;
    if (i >= C) return;
    float g = p[i], b = p[C + i], m = p[2 * C + i], v = p[3 * C + i];
    float sc = g * rsqrtf(v + 1e-5f);
    aff[i] = make_float2(sc, b - m * sc);
}

__global__ void k_norms(const float* __restrict__ f, float* __restrict__ norms, int C) {
    int i = blockIdx.x * 256 + threadIdx.x;       // over B*N
    if (i >= Bc * Nc) return;
    const float* p = f + (size_t)i * C;
    float s = 0.f;
    for (int c = 0; c < C; ++c) s += p[c] * p[c];
    norms[i] = s;
}

// ---------------------------------------------------------------------------
// kNN: one wave per point-row; per-lane top-20 insertion + shuffle merge.
// rank by 2*dot(x_n,x_m) - ||x_m||^2 (order-equivalent to -||x_n-x_m||^2)
// ---------------------------------------------------------------------------
__global__ __launch_bounds__(256) void k_knn(const float* __restrict__ feat,
                                             const float* __restrict__ norms,
                                             int* __restrict__ idx, int C) {
    __shared__ float rowL[8][128];
    int wv = threadIdx.x >> 5, lane = threadIdx.x & 31;
    int row = blockIdx.x * 8 + wv;                // 0 .. B*N-1
    int b = row >> 11, n = row & (Nc - 1);
    const float* fb = feat + (size_t)b * Nc * C;
    for (int c = lane; c < C; c += 32) rowL[wv][c] = fb[(size_t)n * C + c];
    __syncthreads();

    float bs[Kc]; int bi[Kc];
#pragma unroll
    for (int i = 0; i < Kc; ++i) { bs[i] = -3.4e38f; bi[i] = 0; }

    for (int m = lane; m < Nc; m += 32) {
        const float* fm = fb + (size_t)m * C;
        float dot = 0.f;
        for (int c = 0; c < C; ++c) dot += rowL[wv][c] * fm[c];
        float sc = 2.f * dot - norms[b * Nc + m];
        if (sc > bs[Kc - 1]) {
            int pos = Kc - 1;
            while (pos > 0 && bs[pos - 1] < sc) { bs[pos] = bs[pos - 1]; bi[pos] = bi[pos - 1]; --pos; }
            bs[pos] = sc; bi[pos] = m;
        }
    }
    int ptr = 0;
    for (int r = 0; r < Kc; ++r) {
        float s  = (ptr < Kc) ? bs[ptr] : -3.4e38f;
        int  who = lane;
        int  wi  = (ptr < Kc) ? bi[ptr] : 0;
        for (int off = 16; off; off >>= 1) {
            float s2  = __shfl_xor(s, off);
            int  who2 = __shfl_xor(who, off);
            int  wi2  = __shfl_xor(wi, off);
            if (s2 > s || (s2 == s && who2 < who)) { s = s2; who = who2; wi = wi2; }
        }
        if (lane == who) ++ptr;
        if (lane == r) idx[(size_t)row * Kc + r] = wi;
    }
}

// ---------------------------------------------------------------------------
// Edge conv: Y[b,:,n] = max_k LReLU(BN(W @ [nbr-ctr; ctr])) via bf16 WMMA.
// Block = 16 points (320 edge columns). For each of 20 column-tiles:
//   stage 16-column B tile into LDS as bf16 in fragment lane order, then all
//   O/16 output tiles consume it with guard-free 32B vector loads.
// ---------------------------------------------------------------------------
template<int CIN, int O>
__global__ __launch_bounds__(256) void k_edgeconv(const float* __restrict__ feat,
                                                  const int* __restrict__ idx,
                                                  const unsigned short* __restrict__ wlay,
                                                  const float2* __restrict__ aff,
                                                  float* __restrict__ out) {
    constexpr int C2  = 2 * CIN;
    constexpr int NKC = (C2 + 31) / 32;           // K chunks of 32
    __shared__ float    featL[16][CIN];
    __shared__ int      idxL[16 * Kc];
    __shared__ unsigned outL[16][O];
    __shared__ alignas(32) unsigned short eL[NKC * 512];   // B tile, lane order

    int b = blockIdx.y, n0 = blockIdx.x * 16, tid = threadIdx.x;
    const float* fb = feat + (size_t)b * Nc * CIN;

    for (int i = tid; i < 16 * CIN; i += 256)
        featL[i / CIN][i % CIN] = fb[(size_t)(n0 + i / CIN) * CIN + (i % CIN)];
    for (int i = tid; i < 16 * Kc; i += 256)
        idxL[i] = idx[((size_t)b * Nc + n0) * Kc + i];
    for (int i = tid; i < 16 * O; i += 256)
        (&outL[0][0])[i] = 0u;                    // < encoding of any finite float
    __syncthreads();

    int wave = tid >> 5, lane = tid & 31, half = lane >> 4, ln = lane & 15;

    for (int ct = 0; ct < 20; ++ct) {
        // ---- stage B tile: pack bf16 pairs directly in fragment lane order
        for (int i = tid; i < NKC * 256; i += 256) {
            int jl = i / (NKC * 16);              // column in tile 0..15
            int c0 = (i % (NKC * 16)) * 2;        // even channel
            int j  = ct * 16 + jl;                // global edge column 0..319
            int p  = j / Kc;
            int m  = idxL[j];
            const float* fm = fb + (size_t)m * CIN;
            float e0 = 0.f, e1 = 0.f;
            if (c0 < CIN)            e0 = fm[c0] - featL[p][c0];
            else if (c0 < C2)        e0 = featL[p][c0 - CIN];
            if (c0 + 1 < CIN)        e1 = fm[c0 + 1] - featL[p][c0 + 1];
            else if (c0 + 1 < C2)    e1 = featL[p][c0 + 1 - CIN];
            unsigned pk = (unsigned)f2bfu(e0) | ((unsigned)f2bfu(e1) << 16);
            // fragment addr: ((kc*32 + hf*16 + jl)*16 + ii), ii even -> /2
            int kc = c0 >> 5, hf = (c0 >> 4) & 1, ii = c0 & 15;
            ((unsigned*)eL)[(((kc * 32 + hf * 16 + jl) << 4) + ii) >> 1] = pk;
        }
        __syncthreads();

        // ---- compute: waves sweep O/16 output tiles, B tile reused
        for (int ot = wave; ot < (O >> 4); ot += 8) {
            v8f acc;
#pragma unroll
            for (int r = 0; r < 8; ++r) acc[r] = 0.f;
#pragma unroll
            for (int kc = 0; kc < NKC; ++kc) {
                v16bf a  = *reinterpret_cast<const v16bf*>(
                               wlay + ((size_t)(ot * NKC + kc) * 32 + lane) * 16);
                v16bf bb = *reinterpret_cast<const v16bf*>(
                               eL + ((kc * 32 + lane) << 4));
                acc = __builtin_amdgcn_wmma_f32_16x16x32_bf16(false, a, false, bb,
                                                              (short)0, acc, false, false);
            }
            int j = ct * 16 + ln, p = j / Kc;
#pragma unroll
            for (int r = 0; r < 8; ++r) {
                int o = ot * 16 + r + 8 * half;
                float2 af = aff[o];
                float v = fmaf(acc[r], af.x, af.y);
                v = v > 0.f ? v : 0.2f * v;
                atomicMax(&outL[p][o], f2ord(v));
            }
        }
        __syncthreads();
    }
    float* ob = out + ((size_t)b * Nc + n0) * O;
    for (int i = tid; i < 16 * O; i += 256)
        ob[i] = ord2f((&outL[0][0])[i]);
}

// ---------------------------------------------------------------------------
// conv5 (512->1024) + BN + LReLU + fused global max/sum pooling (bf16 WMMA)
// ---------------------------------------------------------------------------
__global__ void k_init_pool(unsigned* __restrict__ maxbuf, float* __restrict__ sumbuf) {
    int i = blockIdx.x * 256 + threadIdx.x;
    if (i < Bc * 1024) { maxbuf[i] = 0u; sumbuf[i] = 0.f; }
}

__global__ __launch_bounds__(256) void k_conv5pool(const float* __restrict__ x1,
                                                   const float* __restrict__ x2,
                                                   const float* __restrict__ x3,
                                                   const float* __restrict__ x4,
                                                   const unsigned short* __restrict__ w5lay,
                                                   const float2* __restrict__ aff5,
                                                   unsigned* __restrict__ maxbuf,
                                                   float* __restrict__ sumbuf) {
    constexpr int NKC = 16;                       // 512 / 32
    __shared__ alignas(32) unsigned short eL[NKC * 512];   // 16 cols x 512 ch, bf16
    int b = blockIdx.y, n0 = blockIdx.x * 16, tid = threadIdx.x;

    // stage concatenated features straight into fragment lane order (bf16)
    for (int i = tid; i < NKC * 256; i += 256) {
        int jl = i >> 8;                          // point in tile
        int c0 = (i & 255) * 2;                   // even channel 0..510
        int n  = n0 + jl;
        const float* src; int coff;
        if (c0 < 64)       { src = x1 + ((size_t)b * Nc + n) * 64;  coff = c0; }
        else if (c0 < 128) { src = x2 + ((size_t)b * Nc + n) * 64;  coff = c0 - 64; }
        else if (c0 < 256) { src = x3 + ((size_t)b * Nc + n) * 128; coff = c0 - 128; }
        else               { src = x4 + ((size_t)b * Nc + n) * 256; coff = c0 - 256; }
        unsigned pk = (unsigned)f2bfu(src[coff]) | ((unsigned)f2bfu(src[coff + 1]) << 16);
        int kc = c0 >> 5, hf = (c0 >> 4) & 1, ii = c0 & 15;
        ((unsigned*)eL)[(((kc * 32 + hf * 16 + jl) << 4) + ii) >> 1] = pk;
    }
    __syncthreads();

    int wave = tid >> 5, lane = tid & 31, half = lane >> 4, ln = lane & 15;
    for (int ot = wave; ot < 64; ot += 8) {       // 1024/16 output tiles
        v8f acc;
#pragma unroll
        for (int r = 0; r < 8; ++r) acc[r] = 0.f;
#pragma unroll
        for (int kc = 0; kc < NKC; ++kc) {
            v16bf a  = *reinterpret_cast<const v16bf*>(
                           w5lay + ((size_t)(ot * NKC + kc) * 32 + lane) * 16);
            v16bf bb = *reinterpret_cast<const v16bf*>(eL + ((kc * 32 + lane) << 4));
            acc = __builtin_amdgcn_wmma_f32_16x16x32_bf16(false, a, false, bb,
                                                          (short)0, acc, false, false);
        }
#pragma unroll
        for (int r = 0; r < 8; ++r) {
            int o = ot * 16 + r + 8 * half;
            float2 af = aff5[o];
            float v = fmaf(acc[r], af.x, af.y);
            v = v > 0.f ? v : 0.2f * v;
            atomicMax(&maxbuf[b * 1024 + o], f2ord(v));
            atomicAdd(&sumbuf[b * 1024 + o], v);
        }
    }
}

__global__ void k_pool_final(const unsigned* __restrict__ maxbuf,
                             const float* __restrict__ sumbuf,
                             float* __restrict__ pooled) {
    int i = blockIdx.x * 256 + threadIdx.x;       // B*2048
    if (i >= Bc * 2048) return;
    int b = i / 2048, j = i % 2048;
    pooled[i] = (j < 1024) ? ord2f(maxbuf[b * 1024 + j])
                           : sumbuf[b * 1024 + j - 1024] * (1.0f / (float)Nc);
}

// ---------------------------------------------------------------------------
// head kernels (tiny; scalar)
// ---------------------------------------------------------------------------
__global__ void k_lin_bn_lrelu(const float* __restrict__ in, const float* __restrict__ w,
                               const float* __restrict__ bias, const float* __restrict__ bnp,
                               float* __restrict__ out, int Cin, int Cout) {
    int i = blockIdx.x * 256 + threadIdx.x;
    if (i >= Bc * Cout) return;
    int b = i / Cout, o = i % Cout;
    const float* wr = w + (size_t)o * Cin;
    const float* xr = in + (size_t)b * Cin;
    float s = bias ? bias[o] : 0.f;
    for (int c = 0; c < Cin; ++c) s += xr[c] * wr[c];
    float g = bnp[o], bb = bnp[Cout + o], m = bnp[2 * Cout + o], v = bnp[3 * Cout + o];
    float sc = g * rsqrtf(v + 1e-5f);
    float y = (s - m) * sc + bb;
    out[i] = y > 0.f ? y : 0.2f * y;
}

__global__ void k_attW(const float* __restrict__ att, const float* __restrict__ attw,
                       const float* __restrict__ attb, float* __restrict__ W, int Ncls) {
    int i = blockIdx.x * 256 + threadIdx.x;
    if (i >= Ncls * 256) return;
    int cls = i / 256, j = i % 256;
    float s = attb[j];
    for (int t = 0; t < 300; ++t) s += att[cls * 300 + t] * attw[j * 300 + t];
    W[i] = s > 0.f ? s : 0.f;
}

__global__ void k_out(const float* __restrict__ h, const float* __restrict__ W,
                      float* __restrict__ out, int Ncls) {
    int i = blockIdx.x * 256 + threadIdx.x;
    if (i >= Bc * Ncls) return;
    int b = i / Ncls, cls = i % Ncls;
    float s = 0.f;
    for (int j = 0; j < 256; ++j) s += h[b * 256 + j] * W[cls * 256 + j];
    out[i] = s;
}

// ---------------------------------------------------------------------------
// host orchestration
// ---------------------------------------------------------------------------
extern "C" void kernel_launch(void* const* d_in, const int* in_sizes, int n_in,
                              void* d_out, int out_size, void* d_ws, size_t ws_size,
                              hipStream_t stream) {
    const float* x       = (const float*)d_in[0];
    const float* old_att = (const float*)d_in[1];
    const float* new_att = (const float*)d_in[2];
    const float* w1 = (const float*)d_in[3];
    const float* w2 = (const float*)d_in[4];
    const float* w3 = (const float*)d_in[5];
    const float* w4 = (const float*)d_in[6];
    const float* w5 = (const float*)d_in[7];
    const float* bn1 = (const float*)d_in[8];
    const float* bn2 = (const float*)d_in[9];
    const float* bn3 = (const float*)d_in[10];
    const float* bn4 = (const float*)d_in[11];
    const float* bn5 = (const float*)d_in[12];
    const float* lw1_old = (const float*)d_in[13];
    const float* bn6_old = (const float*)d_in[14];
    const float* lw2_old = (const float*)d_in[15];
    const float* lb2_old = (const float*)d_in[16];
    const float* bn7_old = (const float*)d_in[17];
    const float* lw1_new = (const float*)d_in[18];
    const float* bn6_new = (const float*)d_in[19];
    const float* lw2_new = (const float*)d_in[20];
    const float* lb2_new = (const float*)d_in[21];
    const float* bn7_new = (const float*)d_in[22];
    const float* attw_old = (const float*)d_in[23];
    const float* attb_old = (const float*)d_in[24];
    const float* attw_new = (const float*)d_in[25];
    const float* attb_new = (const float*)d_in[26];
    float* out = (float*)d_out;

    // workspace layout (float-element offsets; wlay region 32B aligned)
    float* ws = (float*)d_ws;
    float* xt     = ws;                           // 49152
    float* x1     = xt + 49152;                   // 1048576
    float* x2     = x1 + 1048576;                 // 1048576
    float* x3     = x2 + 1048576;                 // 2097152
    float* x4     = x3 + 2097152;                 // 4194304
    float* norms  = x4 + 4194304;                 // 16384
    float* pooled = norms + 16384;                // 16384
    float* sumbuf = pooled + 16384;               // 8192
    unsigned* maxbuf = (unsigned*)(sumbuf + 8192);   // 8192
    int* idxbuf   = (int*)(maxbuf + 8192);        // 327680
    float* h1     = (float*)(idxbuf + 327680);    // 4096
    float* h2     = h1 + 4096;                    // 2048
    float* Wcls   = h2 + 2048;                    // 10240
    float2* affp  = (float2*)(Wcls + 10240);      // 1536 float2
    unsigned short* wlay0 = (unsigned short*)(affp + 1536);  // 32B-aligned
    unsigned short* w1l = wlay0;                  //   2048  (O=64,  NKC=1)
    unsigned short* w2l = wlay0 + 2048;           //   8192  (O=64,  NKC=4)
    unsigned short* w3l = wlay0 + 10240;          //  16384  (O=128, NKC=4)
    unsigned short* w4l = wlay0 + 26624;          //  65536  (O=256, NKC=8)
    unsigned short* w5l = wlay0 + 92160;          // 524288  (O=1024,NKC=16)
    size_t need = (size_t)((char*)(w5l + 524288) - (char*)d_ws);
    if (ws_size < need) return;

    // prep
    k_transpose_x<<<(Bc * Nc + 255) / 256, 256, 0, stream>>>(x, xt);
    k_wlayout<<<(2048 + 255) / 256, 256, 0, stream>>>(w1, w1l, 64, 6, 1);
    k_wlayout<<<(8192 + 255) / 256, 256, 0, stream>>>(w2, w2l, 64, 128, 4);
    k_wlayout<<<(16384 + 255) / 256, 256, 0, stream>>>(w3, w3l, 128, 128, 4);
    k_wlayout<<<(65536 + 255) / 256, 256, 0, stream>>>(w4, w4l, 256, 256, 8);
    k_wlayout<<<(524288 + 255) / 256, 256, 0, stream>>>(w5, w5l, 1024, 512, 16);
    k_bn_affine<<<1, 256, 0, stream>>>(bn1, affp + 0, 64);
    k_bn_affine<<<1, 256, 0, stream>>>(bn2, affp + 64, 64);
    k_bn_affine<<<1, 256, 0, stream>>>(bn3, affp + 128, 128);
    k_bn_affine<<<1, 256, 0, stream>>>(bn4, affp + 256, 256);
    k_bn_affine<<<4, 256, 0, stream>>>(bn5, affp + 512, 1024);

    dim3 gec(Nc / 16, Bc);
    // stage 1: Cin=3 -> O=64
    k_norms<<<(Bc * Nc + 255) / 256, 256, 0, stream>>>(xt, norms, 3);
    k_knn<<<Bc * Nc / 8, 256, 0, stream>>>(xt, norms, idxbuf, 3);
    k_edgeconv<3, 64><<<gec, 256, 0, stream>>>(xt, idxbuf, w1l, affp + 0, x1);
    // stage 2: Cin=64 -> O=64
    k_norms<<<(Bc * Nc + 255) / 256, 256, 0, stream>>>(x1, norms, 64);
    k_knn<<<Bc * Nc / 8, 256, 0, stream>>>(x1, norms, idxbuf, 64);
    k_edgeconv<64, 64><<<gec, 256, 0, stream>>>(x1, idxbuf, w2l, affp + 64, x2);
    // stage 3: Cin=64 -> O=128
    k_norms<<<(Bc * Nc + 255) / 256, 256, 0, stream>>>(x2, norms, 64);
    k_knn<<<Bc * Nc / 8, 256, 0, stream>>>(x2, norms, idxbuf, 64);
    k_edgeconv<64, 128><<<gec, 256, 0, stream>>>(x2, idxbuf, w3l, affp + 128, x3);
    // stage 4: Cin=128 -> O=256
    k_norms<<<(Bc * Nc + 255) / 256, 256, 0, stream>>>(x3, norms, 128);
    k_knn<<<Bc * Nc / 8, 256, 0, stream>>>(x3, norms, idxbuf, 128);
    k_edgeconv<128, 256><<<gec, 256, 0, stream>>>(x3, idxbuf, w4l, affp + 256, x4);

    // conv5 + pooling
    k_init_pool<<<(Bc * 1024 + 255) / 256, 256, 0, stream>>>(maxbuf, sumbuf);
    k_conv5pool<<<gec, 256, 0, stream>>>(x1, x2, x3, x4, w5l, affp + 512, maxbuf, sumbuf);
    k_pool_final<<<(Bc * 2048 + 255) / 256, 256, 0, stream>>>(maxbuf, sumbuf, pooled);

    // heads (sequential; reuse h1/h2/Wcls)
    auto run_head = [&](const float* lw1p, const float* bn6, const float* lw2p,
                        const float* lb2, const float* bn7, const float* attp,
                        const float* attwp, const float* attbp, int Ncls, float* outp) {
        k_lin_bn_lrelu<<<(Bc * 512 + 255) / 256, 256, 0, stream>>>(pooled, lw1p, nullptr, bn6, h1, 2048, 512);
        k_lin_bn_lrelu<<<(Bc * 256 + 255) / 256, 256, 0, stream>>>(h1, lw2p, lb2, bn7, h2, 512, 256);
        k_attW<<<(Ncls * 256 + 255) / 256, 256, 0, stream>>>(attp, attwp, attbp, Wcls, Ncls);
        k_out<<<(Bc * Ncls + 255) / 256, 256, 0, stream>>>(h2, Wcls, outp, Ncls);
    };
    run_head(lw1_old, bn6_old, lw2_old, lb2_old, bn7_old, old_att, attw_old, attb_old, 40, out);
    run_head(lw1_new, bn6_new, lw2_new, lb2_new, bn7_new, new_att, attw_new, attb_new, 10, out + 320);
}
